// Kernel_12281761695451725822_53472342835843
// MI455X (gfx1250) — matrix operand order repaired and checked by reading, not compiled, not run on hardware
//
#include <hip/hip_runtime.h>

// MI455X / gfx1250: wave32, WMMA bf16 16x16x32 with f32 accumulate.
typedef __attribute__((ext_vector_type(16))) __bf16       v16bf;
typedef __attribute__((ext_vector_type(8)))  float        v8f;
typedef __attribute__((ext_vector_type(8)))  unsigned int v8u;

#define C_    64
#define H_    128
#define W_    128
#define PW    132              // padded row: [-2 .. 129] -> idx p, x-col = p-2
#define ROWSZ (C_ * PW)        // 8448 elements per staged row-set
#define KTOT  320              // K = C*5

__global__ __launch_bounds__(256)
void fused_conv_stencil(const float* __restrict__ xin,
                        const float* __restrict__ w3in,
                        const float* __restrict__ w7,
                        float* __restrict__ out)
{
  const float* __restrict__ x  = (const float*)__builtin_assume_aligned(xin, 16);
  const float* __restrict__ w3 = (const float*)__builtin_assume_aligned(w3in, 16);

  // LDS: 3 fp32 rows (h-1,h,h+1) + packed bf16 hi|lo plane of row h.
  extern __shared__ float sm[];                       // 4*ROWSZ*4 = 135168 B
  unsigned int* __restrict__ bpk = (unsigned int*)(sm + 3 * ROWSZ);

  const int nh = blockIdx.x;                          // n*H + h
  const int n  = nh >> 7;
  const int h  = nh & (H_ - 1);

  const int tid   = threadIdx.x;
  const int lane  = tid & 31;
  const int wv    = tid >> 5;                         // 8 waves
  const int row16 = lane & 15;
  const int hi16  = lane >> 4;                        // lane half (VGPR-layout selector)

  const float* __restrict__ xn = x + (size_t)n * C_ * H_ * W_;

  // Warm L2 for the h+2 row stripe (emits global_prefetch_b8).
  if (tid < C_ && h + 2 < H_)
    __builtin_prefetch(&xn[(tid * H_ + h + 2) * W_], 0, 1);

  // ---- stage rows h-1, h, h+1 (zero padded); pack row h as bf16 hi|lo u32 ----
  // Division-free indexing: 192 rows, wave-strided; lanes stride the row.
  for (int rr = wv; rr < 3 * C_; rr += 8) {
    const int  rsel  = rr >> 6;                       // 0,1,2 -> h-1, h, h+1
    const int  c     = rr & (C_ - 1);
    const int  hr    = h + rsel - 1;
    const bool rowok = (unsigned)hr < (unsigned)H_;
    const float* __restrict__ src = xn + (c * H_ + hr) * W_;
    float* __restrict__ dst = sm + rsel * ROWSZ + c * PW;
    for (int p = lane; p < PW; p += 32) {
      float v = 0.0f;
      if (rowok && p >= 2 && p < 130) v = src[p - 2];
      dst[p] = v;
      if (rsel == 1) {                                // one-time split of the GEMM panel
        __bf16 hb = (__bf16)v;
        __bf16 lb = (__bf16)(v - (float)hb);
        unsigned hu = __builtin_bit_cast(unsigned short, hb);
        unsigned lu = __builtin_bit_cast(unsigned short, lb);
        bpk[c * PW + p] = hu | (lu << 16);
      }
    }
  }
  __syncthreads();

  const float* __restrict__ ldsm1 = sm;               // row h-1
  const float* __restrict__ lds0  = sm + ROWSZ;       // row h (fp32, epilogue)
  const float* __restrict__ ldsp1 = sm + 2 * ROWSZ;   // row h+1

  // ---- implicit GEMM: t3[co,w] = sum_kk w3[co,kk] * xrow[kk/5][w + kk%5] ----
  // M=64 -> 4 tiles, N=128 -> 8 tiles; wave wv owns mt = wv&3, nt base = (wv>>2)*64.
  const int mt     = wv & 3;
  const int ntbase = (wv >> 2) * 64;
  const int co     = mt * 16 + row16;                 // A row for this lane
  const float* __restrict__ arow = w3 + co * KTOT;    // contiguous in kk

  v8f acc[4] = {};

#pragma unroll
  for (int kb = 0; kb < 10; ++kb) {
    // A operand, 16x32 bf16: e=0..15 -> K = 16*hi16 + e
    v16bf Ahi, Alo;
    const float* ap = arow + kb * 32 + hi16 * 16;     // 32B-aligned groups
#pragma unroll
    for (int e = 0; e < 8; ++e) {
      float v = ap[e];
      __bf16 hb = (__bf16)v;
      Ahi[e] = hb; Alo[e] = (__bf16)(v - (float)hb);
    }
#pragma unroll
    for (int e = 0; e < 8; ++e) {
      float v = ap[8 + e];
      __bf16 hb = (__bf16)v;
      Ahi[8 + e] = hb; Alo[8 + e] = (__bf16)(v - (float)hb);
    }

#pragma unroll
    for (int nt = 0; nt < 4; ++nt) {
      const int wc = ntbase + nt * 16 + row16;        // B column (N) for this lane
      // B operand, 32x16 bf16: lanes 0-15 hold K=0..15, lanes 16-31 K=16..31.
      // Panel offsets (kk/5)*PW + kk%5 are folded to compile-time constants for
      // both lane halves and picked with one v_cndmask -- no runtime div-by-5.
      v8u bhw, blw;
#pragma unroll
      for (int j = 0; j < 8; ++j) {
        const int kL  = kb * 32 + 2 * j;              // hi16 = 0 variant
        const int kH  = kL + 16;                      // hi16 = 1 variant
        const int o0L = (kL / 5) * PW + (kL % 5);
        const int o0H = (kH / 5) * PW + (kH % 5);
        const int o1L = ((kL + 1) / 5) * PW + ((kL + 1) % 5);
        const int o1H = ((kH + 1) / 5) * PW + ((kH + 1) % 5);
        unsigned p0 = bpk[wc + (hi16 ? o0H : o0L)];   // {lo|hi} of element 2j
        unsigned p1 = bpk[wc + (hi16 ? o1H : o1L)];   // {lo|hi} of element 2j+1
        bhw[j] = __builtin_amdgcn_perm(p1, p0, 0x05040100u);  // {hi1,hi0}
        blw[j] = __builtin_amdgcn_perm(p1, p0, 0x07060302u);  // {lo1,lo0}
      }
      v16bf Bhi = __builtin_bit_cast(v16bf, bhw);
      v16bf Blo = __builtin_bit_cast(v16bf, blw);

      // split-bf16 product: hi*hi + hi*lo + lo*hi  (error ~2^-16 relative)
      acc[nt] = __builtin_amdgcn_wmma_f32_16x16x32_bf16(
          false, Ahi, false, Bhi, (short)0, acc[nt], false, false);
      acc[nt] = __builtin_amdgcn_wmma_f32_16x16x32_bf16(
          false, Ahi, false, Blo, (short)0, acc[nt], false, false);
      acc[nt] = __builtin_amdgcn_wmma_f32_16x16x32_bf16(
          false, Alo, false, Bhi, (short)0, acc[nt], false, false);
    }
  }

  // ---- epilogue: t7 stencil (exact fp32 from LDS) * t3 (accumulators) ----
  // C/D layout: VGPR r -> M = r + 8*hi16, N = row16.
  float* __restrict__ outn = out + (size_t)n * C_ * H_ * W_;
#pragma unroll
  for (int r = 0; r < 8; ++r) {
    const int c = mt * 16 + r + 8 * hi16;
    float w7r[9];
#pragma unroll
    for (int q = 0; q < 9; ++q) w7r[q] = w7[c * 9 + q];
    const float* r0 = ldsm1 + c * PW;
    const float* r1 = lds0  + c * PW;
    const float* r2 = ldsp1 + c * PW;
    const float* rows[3] = { r0, r1, r2 };
#pragma unroll
    for (int nt = 0; nt < 4; ++nt) {
      const int wc = ntbase + nt * 16 + row16;
      const int wr = (wc + W_ - 1) & (W_ - 1);        // roll(+1): src = (w-1) mod W
      // lhs(b) = x[c,h, w + 2(b-1)]  -> padded idx w + 2b
      float lhsv[3] = { r1[wc], r1[wc + 2], r1[wc + 4] };
      float t7 = 0.0f;
#pragma unroll
      for (int a = 0; a < 3; ++a) {                   // t5 width offset
#pragma unroll
        for (int b = 0; b < 3; ++b) {                 // t5 height offset / t1 width offset
          float rv = rows[b][wr + 2 * a];             // padded idx wr + 2a
          float m  = fmaxf(lhsv[b], rv);
          t7 = fmaf(w7r[a * 3 + b], m, t7);
        }
      }
      outn[(c * H_ + h) * W_ + wc] = t7 * acc[nt][r];
    }
  }
}

extern "C" void kernel_launch(void* const* d_in, const int* in_sizes, int n_in,
                              void* d_out, int out_size, void* d_ws, size_t ws_size,
                              hipStream_t stream) {
  (void)in_sizes; (void)n_in; (void)out_size; (void)d_ws; (void)ws_size;
  const float* x  = (const float*)d_in[0];
  const float* w3 = (const float*)d_in[1];
  const float* w7 = (const float*)d_in[2];
  float* out = (float*)d_out;

  const size_t shmem = (size_t)4 * ROWSZ * 4;         // 135168 B < 320 KB/WGP
  fused_conv_stencil<<<dim3(16 * H_), dim3(256), shmem, stream>>>(x, w3, w7, out);
}
